// NodeUpdateNetwork_61229053772127
// MI455X (gfx1250) — matrix operand order repaired and checked by reading, not compiled, not run on hardware
//
#include <hip/hip_runtime.h>
#include <hip/hip_bf16.h>
#include <cstdint>
#include <cstddef>

// GatedGCN layer for MI455X (gfx1250): bf16 WMMA GEMMs, f32 accumulate.
// N=50000 nodes, E=800000 edges, F=96. dst = repeat(arange(N), 16), so each
// node's incoming edges are contiguous (16 per node).

#define F 96
#define DEG 16
#define WPITCH 104   // bf16 elems per weight row in LDS: 208B rows (16B aligned, conflict-free)
#define E1PITCH 100  // f32 elems per e1 staging row: 400B (16B aligned, conflict-free)

typedef __attribute__((ext_vector_type(16))) __bf16 v16bf;
typedef __attribute__((ext_vector_type(8)))  float  v8f;

union BfFrag { v16bf v; unsigned int u[8]; uint4 q[2]; };

__device__ __forceinline__ unsigned short f2bf(float a) {
    union { __bf16 h; unsigned short u; } c;
    c.h = (__bf16)a;                      // hardware v_cvt_*bf16_f32
    return c.u;
}

// A fragment (16x32 bf16, M x K) for K-chunk c, built from a per-lane f32 row
// with hardware cvt. Lane holds K = c*32 + ksel*8 + {0..7} and +16 more.
__device__ __forceinline__ v16bf a_frag_f32(const float* row, int c, int ksel) {
    const float4* p0 = (const float4*)(row + c * 32 + ksel * 8);
    const float4* p1 = (const float4*)(row + c * 32 + ksel * 8 + 16);
    float4 x0 = p0[0], x1 = p0[1];
    float4 y0 = p1[0], y1 = p1[1];
    v16bf v;
    v[0]  = (__bf16)x0.x; v[1]  = (__bf16)x0.y; v[2]  = (__bf16)x0.z; v[3]  = (__bf16)x0.w;
    v[4]  = (__bf16)x1.x; v[5]  = (__bf16)x1.y; v[6]  = (__bf16)x1.z; v[7]  = (__bf16)x1.w;
    v[8]  = (__bf16)y0.x; v[9]  = (__bf16)y0.y; v[10] = (__bf16)y0.z; v[11] = (__bf16)y0.w;
    v[12] = (__bf16)y1.x; v[13] = (__bf16)y1.y; v[14] = (__bf16)y1.z; v[15] = (__bf16)y1.w;
    return v;
}

// A fragment straight from a pre-converted bf16 row: two b128 loads, no VALU.
__device__ __forceinline__ v16bf a_frag_bf(const unsigned short* row, int c, int ksel) {
    BfFrag f;
    f.q[0] = *(const uint4*)(row + c * 32 + ksel * 8);
    f.q[1] = *(const uint4*)(row + c * 32 + ksel * 8 + 16);
    return f.v;
}

// B fragment (32x16 bf16, K x N) from LDS bf16 weights [rows][WPITCH] row-major.
// out[m][n] = sum_k x[m][k]*W[n][k]  =>  B[k][n] = W[wrow][k], wrow = n_global.
// Lane holds K = kc*32 + (lane>>4)*8 + {0..7} and +16 more.
__device__ __forceinline__ v16bf b_frag(const unsigned short* w, int wrow, int kc, int ksel) {
    const unsigned short* p = w + wrow * WPITCH + kc * 32 + ksel * 8;
    BfFrag f;
    f.q[0] = *(const uint4*)(p);
    f.q[1] = *(const uint4*)(p + 16);
    return f.v;
}

__device__ __forceinline__ v8f wmma_bf16(v16bf a, v16bf b, v8f c) {
    return __builtin_amdgcn_wmma_f32_16x16x32_bf16(false, a, false, b, (short)0, c, false, false);
}

__device__ __forceinline__ float fast_rcp(float x) { return __builtin_amdgcn_rcpf(x); }
__device__ __forceinline__ float sigmoidf_(float x) { return fast_rcp(1.0f + __expf(-x)); }
__device__ __forceinline__ float tanhf_(float x) {
    float ax = fabsf(x);
    float t = __expf(-2.0f * ax);                     // (0,1], no overflow
    float r = 1.0f - 2.0f * t * fast_rcp(1.0f + t);   // tanh(|x|)
    return copysignf(r, x);
}
__device__ __forceinline__ v8f splat8(float b) { v8f v = {b, b, b, b, b, b, b, b}; return v; }

// ---------------------------------------------------------------------------
// One 16-row GRU cell tile: outv = (1-z)*tanh(i_n + r*h_n) + z*hprev
// ax/ah: bf16 A-fragments (3 K-chunks). Weights in LDS bf16. Gates processed
// in phases (r -> n -> z) to bound register pressure. Result stays in C-layout
// registers: lane l, reg g -> (m = g + 8*(l>>4), n = nt*16 + (l&15)).
// FIRST: hprev read from a global f32 tile (pitch F); else from registers.
// ---------------------------------------------------------------------------
template <bool FIRST>
__device__ __forceinline__ void gru_tile(
    const v16bf ax[3], const v16bf ah[3],
    const unsigned short* wih, const unsigned short* whh,
    const float* bih, const float* bhh,
    const float* hprev, const v8f* hprev_r,
    v8f outv[6], int lm, int ksel)
{
    // r gate: columns [0,96)
    v8f racc[6];
#pragma unroll
    for (int nt = 0; nt < 6; ++nt) {
        int col = nt * 16 + lm;
        v8f a = splat8(bih[col] + bhh[col]);
#pragma unroll
        for (int kc = 0; kc < 3; ++kc) {
            a = wmma_bf16(ax[kc], b_frag(wih, col, kc, ksel), a);
            a = wmma_bf16(ah[kc], b_frag(whh, col, kc, ksel), a);
        }
        racc[nt] = a;
    }
    // n gate: columns [192,288); n = tanh(i_n + r*h_n)
    v8f nval[6];
#pragma unroll
    for (int nt = 0; nt < 6; ++nt) {
        int col = nt * 16 + lm, w = 192 + col;
        v8f ai = splat8(bih[w]);
        v8f an = splat8(bhh[w]);
#pragma unroll
        for (int kc = 0; kc < 3; ++kc) {
            ai = wmma_bf16(ax[kc], b_frag(wih, w, kc, ksel), ai);
            an = wmma_bf16(ah[kc], b_frag(whh, w, kc, ksel), an);
        }
#pragma unroll
        for (int g = 0; g < 8; ++g) {
            float r = sigmoidf_(racc[nt][g]);
            nval[nt][g] = tanhf_(ai[g] + r * an[g]);
        }
    }
    // z gate: columns [96,192); blend with hprev
#pragma unroll
    for (int nt = 0; nt < 6; ++nt) {
        int col = nt * 16 + lm, w = 96 + col;
        v8f a = splat8(bih[w] + bhh[w]);
#pragma unroll
        for (int kc = 0; kc < 3; ++kc) {
            a = wmma_bf16(ax[kc], b_frag(wih, w, kc, ksel), a);
            a = wmma_bf16(ah[kc], b_frag(whh, w, kc, ksel), a);
        }
#pragma unroll
        for (int g = 0; g < 8; ++g) {
            int m = g + 8 * ksel;
            float z = sigmoidf_(a[g]);
            float hp = FIRST ? hprev[m * F + col] : hprev_r[nt][g];
            outv[nt][g] = (1.0f - z) * nval[nt][g] + z * hp;
        }
    }
}

// ---------------------------------------------------------------------------
// Kernel 0: zero the BN-stats accumulators (4*96 floats)
// ---------------------------------------------------------------------------
__global__ void k_init_stats(float* stats) {
    int i = blockIdx.x * blockDim.x + threadIdx.x;
    if (i < 4 * F) stats[i] = 0.0f;
}

// ---------------------------------------------------------------------------
// Kernel 1: Ah = x@A_w^T + A_b, Bh = x@B_w^T + B_b via WMMA.
// One wave per 16-node tile, grid-strided. Weights bf16 in LDS.
// ---------------------------------------------------------------------------
__global__ __launch_bounds__(256) void k_node_gemm(
    const float* __restrict__ node_feat,
    const float* __restrict__ A_w, const float* __restrict__ A_b,
    const float* __restrict__ B_w, const float* __restrict__ B_b,
    float* __restrict__ Ah, float* __restrict__ Bh, int nTiles)
{
    __shared__ unsigned short wa[F * WPITCH];
    __shared__ unsigned short wb[F * WPITCH];
    int tid = threadIdx.x;
    for (int i = tid; i < F * F; i += 256) {
        int r = i / F, c = i - r * F;
        wa[r * WPITCH + c] = f2bf(A_w[i]);
        wb[r * WPITCH + c] = f2bf(B_w[i]);
    }
    __syncthreads();
    int lane = tid & 31, wv = tid >> 5;
    int lm = lane & 15, ksel = lane >> 4;
    for (int t = blockIdx.x * 8 + wv; t < nTiles; t += gridDim.x * 8) {
        const float* xrow = node_feat + (size_t)(t * 16 + lm) * F;
        v16bf ax[3];
#pragma unroll
        for (int c = 0; c < 3; ++c) ax[c] = a_frag_f32(xrow, c, ksel);
#pragma unroll
        for (int mtx = 0; mtx < 2; ++mtx) {
            const unsigned short* w = mtx ? wb : wa;
            const float* bias = mtx ? B_b : A_b;
            float* out = mtx ? Bh : Ah;
#pragma unroll
            for (int nt = 0; nt < 6; ++nt) {
                int col = nt * 16 + lm;
                v8f acc = splat8(bias[col]);
#pragma unroll
                for (int kc = 0; kc < 3; ++kc)
                    acc = wmma_bf16(ax[kc], b_frag(w, col, kc, ksel), acc);
#pragma unroll
                for (int g = 0; g < 8; ++g) {
                    int m = g + 8 * ksel;
                    out[(size_t)(t * 16 + m) * F + col] = acc[g];
                }
            }
        }
    }
}

// ---------------------------------------------------------------------------
// Kernel 2: edge-gated mean aggregation. One wave per node (its 16 edges are
// contiguous). h = Ah + sum(sig(e)*Bh[src]) / (sum(sig(e)) + eps).
// Also: BN partial sums for h (f32 atomics) and a bf16 mirror of h so the GRU
// kernel's h[src]/h[dst] A-fragments are pure b128 loads (half the gather B/W).
// ---------------------------------------------------------------------------
__global__ __launch_bounds__(256) void k_aggregate(
    const float* __restrict__ edge_feat, const int* __restrict__ src,
    const float* __restrict__ Ah, const float* __restrict__ Bh,
    float* __restrict__ h, unsigned short* __restrict__ h_bf,
    float* __restrict__ stats, int nNodes)
{
    int tid = threadIdx.x, lane = tid & 31, wv = tid >> 5;
    float s0 = 0, s1 = 0, s2 = 0, q0 = 0, q1 = 0, q2 = 0;
    for (int node = blockIdx.x * 8 + wv; node < nNodes; node += gridDim.x * 8) {
        float num0 = 0, num1 = 0, num2 = 0, den0 = 0, den1 = 0, den2 = 0;
        size_t ebase = (size_t)node * DEG;
#pragma unroll 4
        for (int k = 0; k < DEG; ++k) {
            size_t e = ebase + k;
            int sj = src[e];
            const float* er = edge_feat + e * F;
            const float* br = Bh + (size_t)sj * F;
            float g0 = sigmoidf_(er[lane]);
            float g1 = sigmoidf_(er[lane + 32]);
            float g2 = sigmoidf_(er[lane + 64]);
            num0 += g0 * br[lane];      den0 += g0;
            num1 += g1 * br[lane + 32]; den1 += g1;
            num2 += g2 * br[lane + 64]; den2 += g2;
        }
        const float* ar = Ah + (size_t)node * F;
        float h0 = ar[lane]      + num0 / (den0 + 1e-6f);
        float h1 = ar[lane + 32] + num1 / (den1 + 1e-6f);
        float h2 = ar[lane + 64] + num2 / (den2 + 1e-6f);
        float* hr = h + (size_t)node * F;
        hr[lane] = h0; hr[lane + 32] = h1; hr[lane + 64] = h2;
        unsigned short* hb = h_bf + (size_t)node * F;
        hb[lane] = f2bf(h0); hb[lane + 32] = f2bf(h1); hb[lane + 64] = f2bf(h2);
        s0 += h0; s1 += h1; s2 += h2;
        q0 += h0 * h0; q1 += h1 * h1; q2 += h2 * h2;
    }
    atomicAdd(&stats[lane], s0);
    atomicAdd(&stats[lane + 32], s1);
    atomicAdd(&stats[lane + 64], s2);
    atomicAdd(&stats[96 + lane], q0);
    atomicAdd(&stats[96 + lane + 32], q1);
    atomicAdd(&stats[96 + lane + 64], q2);
}

// ---------------------------------------------------------------------------
// Kernel 3: fused edge GRU chain: e' = GRU2(h[dst], GRU1(h[src], e)).
// All four 288x96 weight matrices resident in LDS as bf16 (~240 KB) plus a
// per-wave f32 staging tile for e1 (50 KB): ~291 KB of the 320 KB WGP LDS.
// One wave per 16-edge tile, grid-strided; 216 WMMAs per tile. GRU1's output
// tile stays in registers as GRU2's hprev; it also round-trips through LDS
// once for the cross-lane A-fragment transpose. Writes GRU2 output (pre-BN)
// directly into d_out's e region + BN partials.
// ---------------------------------------------------------------------------
__global__ __launch_bounds__(256) void k_gru(
    const float* __restrict__ edge_feat,
    const int* __restrict__ src, const int* __restrict__ dst,
    const unsigned short* __restrict__ h_bf,
    const float* __restrict__ w1ih_g, const float* __restrict__ w1hh_g,
    const float* __restrict__ b1ih, const float* __restrict__ b1hh,
    const float* __restrict__ w2ih_g, const float* __restrict__ w2hh_g,
    const float* __restrict__ b2ih, const float* __restrict__ b2hh,
    float* __restrict__ e_out, float* __restrict__ stats, int nTiles)
{
    __shared__ unsigned short w1ih[3 * F * WPITCH];
    __shared__ unsigned short w1hh[3 * F * WPITCH];
    __shared__ unsigned short w2ih[3 * F * WPITCH];
    __shared__ unsigned short w2hh[3 * F * WPITCH];
    __shared__ float e1buf[8 * 16 * E1PITCH];

    int tid = threadIdx.x;
    for (int i = tid; i < 3 * F * F; i += 256) {
        int r = i / F, c = i - r * F;
        int o = r * WPITCH + c;
        w1ih[o] = f2bf(w1ih_g[i]);
        w1hh[o] = f2bf(w1hh_g[i]);
        w2ih[o] = f2bf(w2ih_g[i]);
        w2hh[o] = f2bf(w2hh_g[i]);
    }
    __syncthreads();

    int lane = tid & 31, wv = tid >> 5;
    int lm = lane & 15, ksel = lane >> 4;
    float* e1 = e1buf + wv * (16 * E1PITCH);
    float s[6] = {0, 0, 0, 0, 0, 0}, q[6] = {0, 0, 0, 0, 0, 0};
    int wstride = gridDim.x * 8;

    for (int t = blockIdx.x * 8 + wv; t < nTiles; t += wstride) {
        size_t ebase = (size_t)t * 16;
        // Prefetch next tile's edge row for this lane (global_prefetch_b8).
        if (t + wstride < nTiles)
            __builtin_prefetch(edge_feat + ((size_t)(t + wstride) * 16 + lm) * F, 0, 3);

        v16bf ax[3], ah[3];
        {
            const unsigned short* xr = h_bf + (size_t)src[ebase + lm] * F; // h[src] gather (bf16, L2)
            const float* er = edge_feat + (ebase + lm) * F;
#pragma unroll
            for (int c = 0; c < 3; ++c) { ax[c] = a_frag_bf(xr, c, ksel); ah[c] = a_frag_f32(er, c, ksel); }
        }
        // e1 = GRU1(h[src], edge_feat); hprev = edge_feat (f32, global)
        v8f o1[6];
        gru_tile<true>(ax, ah, w1ih, w1hh, b1ih, b1hh,
                       edge_feat + ebase * F, nullptr, o1, lm, ksel);
        // Stage e1 in LDS (f32) for the cross-lane A-fragment transpose.
#pragma unroll
        for (int nt = 0; nt < 6; ++nt)
#pragma unroll
            for (int g = 0; g < 8; ++g)
                e1[(g + 8 * ksel) * E1PITCH + nt * 16 + lm] = o1[nt][g];
        {
            const unsigned short* xr = h_bf + (size_t)dst[ebase + lm] * F; // h[dst] gather (bf16)
            const float* er = e1 + lm * E1PITCH;
#pragma unroll
            for (int c = 0; c < 3; ++c) { ax[c] = a_frag_bf(xr, c, ksel); ah[c] = a_frag_f32(er, c, ksel); }
        }
        // e = GRU2(h[dst], e1); hprev = e1 (straight from registers)
        v8f o2[6];
        gru_tile<false>(ax, ah, w2ih, w2hh, b2ih, b2hh, nullptr, o1, o2, lm, ksel);
        // Emit pre-BN e and BN partials.
        float* orow = e_out + ebase * F;
#pragma unroll
        for (int nt = 0; nt < 6; ++nt) {
            int col = nt * 16 + lm;
#pragma unroll
            for (int g = 0; g < 8; ++g) {
                float o = o2[nt][g];
                orow[(g + 8 * ksel) * F + col] = o;
                s[nt] += o; q[nt] += o * o;
            }
        }
    }
#pragma unroll
    for (int nt = 0; nt < 6; ++nt) {
        int col = nt * 16 + lm;
        atomicAdd(&stats[col], s[nt]);
        atomicAdd(&stats[96 + col], q[nt]);
    }
}

// ---------------------------------------------------------------------------
// Kernel 4: BatchNorm (biased batch stats) -> ReLU -> residual, for h and e.
// e is transformed in place inside d_out.
// ---------------------------------------------------------------------------
__global__ __launch_bounds__(256) void k_finalize(
    const float* __restrict__ node_feat, const float* __restrict__ edge_feat,
    const float* __restrict__ h, const float* __restrict__ stats,
    const float* __restrict__ gh, const float* __restrict__ bh,
    const float* __restrict__ ge, const float* __restrict__ be,
    float* __restrict__ out_h, float* __restrict__ out_e,
    int nNodes, long long nEdges)
{
    size_t nh = (size_t)nNodes * F;
    size_t ne = (size_t)nEdges * F;
    size_t total = nh + ne;
    float invN = 1.0f / (float)nNodes;
    float invE = 1.0f / (float)nEdges;
    for (size_t i = (size_t)blockIdx.x * blockDim.x + threadIdx.x; i < total;
         i += (size_t)gridDim.x * blockDim.x) {
        if (i < nh) {
            int col = (int)(i % F);
            float m = stats[col] * invN;
            float v = stats[96 + col] * invN - m * m;
            float y = (h[i] - m) * rsqrtf(v + 1e-5f) * gh[col] + bh[col];
            out_h[i] = node_feat[i] + fmaxf(y, 0.0f);
        } else {
            size_t j = i - nh;
            int col = (int)(j % F);
            float m = stats[192 + col] * invE;
            float v = stats[288 + col] * invE - m * m;
            float y = (out_e[j] - m) * rsqrtf(v + 1e-5f) * ge[col] + be[col];
            out_e[j] = edge_feat[j] + fmaxf(y, 0.0f);
        }
    }
}

// ---------------------------------------------------------------------------
extern "C" void kernel_launch(void* const* d_in, const int* in_sizes, int n_in,
                              void* d_out, int out_size, void* d_ws, size_t ws_size,
                              hipStream_t stream)
{
    const float* node_feat = (const float*)d_in[0];
    const float* edge_feat = (const float*)d_in[1];
    const int*   src       = (const int*)d_in[2];
    const int*   dst       = (const int*)d_in[3];
    const float* A_w   = (const float*)d_in[4];
    const float* A_b   = (const float*)d_in[5];
    const float* B_w   = (const float*)d_in[6];
    const float* B_b   = (const float*)d_in[7];
    const float* g1wih = (const float*)d_in[8];
    const float* g1whh = (const float*)d_in[9];
    const float* g1bih = (const float*)d_in[10];
    const float* g1bhh = (const float*)d_in[11];
    const float* g2wih = (const float*)d_in[12];
    const float* g2whh = (const float*)d_in[13];
    const float* g2bih = (const float*)d_in[14];
    const float* g2bhh = (const float*)d_in[15];
    const float* bn_h_g = (const float*)d_in[16];
    const float* bn_h_b = (const float*)d_in[17];
    const float* bn_e_g = (const float*)d_in[18];
    const float* bn_e_b = (const float*)d_in[19];

    int nNodes = in_sizes[0] / F;                  // 50000
    long long nEdges = (long long)in_sizes[1] / F; // 800000

    // Workspace layout: Ah | Bh | h (f32) | stats[4*96] | h_bf (bf16)
    float* ws = (float*)d_ws;
    float* Ah = ws;
    float* Bh = Ah + (size_t)nNodes * F;
    float* hN = Bh + (size_t)nNodes * F;
    float* stats = hN + (size_t)nNodes * F;
    unsigned short* h_bf = (unsigned short*)(stats + 4 * F);

    float* out_h = (float*)d_out;
    float* out_e = out_h + (size_t)nNodes * F;

    k_init_stats<<<2, 256, 0, stream>>>(stats);
    k_node_gemm<<<256, 256, 0, stream>>>(node_feat, A_w, A_b, B_w, B_b, Ah, Bh, nNodes / 16);
    k_aggregate<<<512, 256, 0, stream>>>(edge_feat, src, Ah, Bh, hN, h_bf, stats, nNodes);
    k_gru<<<512, 256, 0, stream>>>(edge_feat, src, dst, h_bf,
                                   g1wih, g1whh, g1bih, g1bhh,
                                   g2wih, g2whh, g2bih, g2bhh,
                                   out_e, stats + 192, (int)(nEdges / 16));
    k_finalize<<<1024, 256, 0, stream>>>(node_feat, edge_feat, hN, stats,
                                         bn_h_g, bn_h_b, bn_e_g, bn_e_b,
                                         out_h, out_e, nNodes, nEdges);
}
